// RLSTM_Model_4836133175331
// MI455X (gfx1250) — matrix operand order repaired and checked by reading, not compiled, not run on hardware
//
#include <hip/hip_runtime.h>
#include <stdint.h>

typedef __attribute__((ext_vector_type(16))) __bf16 v16bf;
typedef __attribute__((ext_vector_type(8)))  __bf16 v8bf;
typedef __attribute__((ext_vector_type(8)))  float  v8f;

#define T_STEPS 1024
#define BATCH   32
#define FDIM    256
#define HDIM    256
#define GCOLS   768   // 3*H: o-gate chunk is dead (o_t = i_t in the source)
#define NT_G    48    // GCOLS / 16
#define KSTEPS  8     // 256 / 32 (bf16 WMMA K=32)
#define NTILES  96    // 48 ntiles * 2 mtiles (gate matrix tiles); tile = nt*2+mtile
#define PAX_ELTS_PER_T ((size_t)NTILES * 32 * 8)   // tile-packed fragment order

// ---------- scalar helpers (native bf16 converts on gfx1250) ----------
static __device__ __forceinline__ __bf16 f2bf(float f) { return (__bf16)f; }
static __device__ __forceinline__ float  bf2f(__bf16 b) { return (float)b; }
static __device__ __forceinline__ float sigmoidf_(float x) { return 1.f / (1.f + __expf(-x)); }
static __device__ __forceinline__ float tanhf_(float x)    { return 1.f - 2.f / (__expf(2.f * x) + 1.f); }

// ---------- CDNA5 async global<->LDS bulk copies (ASYNCcnt-tracked) ----------
static __device__ __forceinline__ void async_load_b128_to_lds(const void* gaddr, const void* lds) {
    uint32_t l = (uint32_t)(uintptr_t)lds;         // flat shared addr low bits = LDS byte address
    uint64_t g = (uint64_t)(uintptr_t)gaddr;
    asm volatile("global_load_async_to_lds_b128 %0, %1, off" :: "v"(l), "v"(g) : "memory");
}
static __device__ __forceinline__ void async_store_b128_from_lds(void* gaddr, const void* lds) {
    uint32_t l = (uint32_t)(uintptr_t)lds;
    uint64_t g = (uint64_t)(uintptr_t)gaddr;
    asm volatile("global_store_async_from_lds_b128 %0, %1, off" :: "v"(g), "v"(l) : "memory");
}
static __device__ __forceinline__ void wait_async0() {
    asm volatile("s_wait_asynccnt 0x0" ::: "memory");
}
static __device__ __forceinline__ void wait_tensor0() {
#if __has_builtin(__builtin_amdgcn_s_wait_tensorcnt)
    __builtin_amdgcn_s_wait_tensorcnt((short)0);
#else
    asm volatile("s_wait_tensorcnt 0x0" ::: "memory");
#endif
}

// ---------- Tensor Data Mover (TDM) descriptor path ----------
#if __has_builtin(__builtin_amdgcn_tensor_load_to_lds)
#define HAVE_TDM 1
typedef unsigned int uint32x4 __attribute__((ext_vector_type(4)));
typedef int          int32x4  __attribute__((ext_vector_type(4)));
typedef int          int32x8  __attribute__((ext_vector_type(8)));

// 1D copy of `bytes` (multiple of 8, <= 8B*65535) from global to LDS via TDM.
// Must be called from one wave; wave-uniform args enforced via readfirstlane.
static __device__ __forceinline__ void tdm_load_1d(const void* gaddr, const void* lds, uint32_t bytes) {
    uint32_t n8    = bytes >> 3;                                   // 8-byte elements
    uint32_t laddr = (uint32_t)__builtin_amdgcn_readfirstlane((int)(uint32_t)(uintptr_t)lds);
    uint64_t ga    = (uint64_t)(uintptr_t)gaddr;
    uint32_t ga_lo = (uint32_t)__builtin_amdgcn_readfirstlane((int)(uint32_t)ga);
    uint32_t ga_hi = (uint32_t)__builtin_amdgcn_readfirstlane((int)(uint32_t)(ga >> 32));
    // D# group 0: count=1 | lds_addr | global_addr[56:0] | type=2 ("image")
    uint32x4 g0 = { 1u, laddr, ga_lo, (ga_hi & 0x01FFFFFFu) | (2u << 30) };
    // D# group 1: data_size=3 (8B); tensor_dim0=n8; tensor_dim1=1; tile_dim0=n8;
    //             tile_dim1=1; tensor_dim0_stride=n8
    int32x8 g1 = { (int)(3u << 16), (int)(n8 << 16), (int)(1u << 16), (int)(n8 << 16),
                   1, (int)n8, 0, 0 };
    int32x4 z4 = { 0, 0, 0, 0 };
#if defined(__clang_major__) && (__clang_major__ >= 23)
    int32x8 z8 = { 0, 0, 0, 0, 0, 0, 0, 0 };
    __builtin_amdgcn_tensor_load_to_lds(g0, g1, z4, z4, z8, 0);
#else
    __builtin_amdgcn_tensor_load_to_lds(g0, g1, z4, z4, 0);
#endif
}
#endif

// ---------- WMMA fragment helpers ----------
static __device__ __forceinline__ v8f wmma_bf16(v16bf a, v16bf b, v8f c) {
    return __builtin_amdgcn_wmma_f32_16x16x32_bf16(false, a, false, b, (short)0, c, false, false);
}

// A-fragment (16x32 bf16) from row-major [rows][ld] source, contiguous K order (same as B):
// lanes 0-15: M=lane, K={0..15}; lanes 16-31: M=lane-16, K={16..31}
static __device__ __forceinline__ v16bf load_A(const __bf16* base, int row0, int ld, int k0, int lane) {
    int m   = row0 + (lane & 15);
    int sel = lane >> 4;
    const __bf16* r = base + (size_t)m * ld + k0;
    v8bf lo = *(const v8bf*)(r + sel * 16);
    v8bf hi = *(const v8bf*)(r + sel * 16 + 8);
    v16bf a;
#pragma unroll
    for (int i = 0; i < 8; ++i) { a[i] = lo[i]; a[i + 8] = hi[i]; }
    return a;
}

// B-fragment (32x16 bf16) from K-contiguous "transposed" source src[n][ldk]:
// lane -> column n0+(lane&15); 16 contiguous K values at (lane>>4)*16.
static __device__ __forceinline__ v16bf load_B_T(const __bf16* src, int n0, int ldk, int lane) {
    const __bf16* r = src + (size_t)(n0 + (lane & 15)) * ldk + (lane >> 4) * 16;
    return *(const v16bf*)r;
}

// B-fragment from pre-packed tiles: one contiguous 32-byte run per lane.
static __device__ __forceinline__ v16bf load_B_pack(const __bf16* pack, int tile, int lane) {
    return *(const v16bf*)(pack + ((size_t)tile * 32 + lane) * 16);
}

// ---------- prep kernels ----------
__global__ void convert_p_kernel(const float* __restrict__ src, __bf16* __restrict__ dst) {
    int i = blockIdx.x * 256 + threadIdx.x;
    if (i < BATCH * 64) dst[i] = f2bf(src[i]);
}

// Pack [256][1024] f32 weight into WMMA B-fragment tile order (only first 768 cols).
__global__ void pack_weight_kernel(const float* __restrict__ src, __bf16* __restrict__ dst) {
    int gid = blockIdx.x * 256 + threadIdx.x;             // KSTEPS*NT_G*32 = 12288
    if (gid >= KSTEPS * NT_G * 32) return;
    int ks = gid / (NT_G * 32);
    int rem = gid % (NT_G * 32);
    int nt = rem / 32, lane = rem % 32;
    int n  = nt * 16 + (lane & 15);
    int kh = lane >> 4;
    __bf16* out = dst + ((size_t)(ks * NT_G + nt) * 32 + lane) * 16;
#pragma unroll
    for (int v = 0; v < 16; ++v) {
        int k = ks * 32 + kh * 16 + v;
        out[v] = f2bf(src[(size_t)k * 1024 + n]);
    }
}

// r_x[t][b] = x[b,t,:] . W_r
__global__ void recall_x_kernel(const float* __restrict__ x, const float* __restrict__ Wr,
                                float* __restrict__ rx) {
    int gid = blockIdx.x * 256 + threadIdx.x;
    if (gid >= T_STEPS * BATCH) return;
    int t = gid >> 5, b = gid & 31;
    const float* xp = x + ((size_t)b * T_STEPS + t) * FDIM;
    float s = 0.f;
    for (int f = 0; f < FDIM; ++f) s += xp[f] * Wr[f];
    rx[t * BATCH + b] = s;
}

// ---------- time-parallel precompute: PAX[t]=P1@x_t@W, PBX[t]=P2@x_t@W ----------
// Output stored in tile-packed C-fragment order: [t][tile(=nt*2+mtile)][lane][8]
__global__ __launch_bounds__(256, 2) void precompute_kernel(
    const float* __restrict__ x, const __bf16* __restrict__ Pbf,
    const __bf16* __restrict__ Wpack,
    __bf16* __restrict__ PAX, __bf16* __restrict__ PBX) {
    __shared__ __align__(64) __bf16 xT[FDIM][BATCH];      // x_t transposed (K-contiguous)
    __shared__ __align__(64) __bf16 Y[2][BATCH][FDIM];    // P1@x_t, P2@x_t
    int t = blockIdx.x;
    int tid = threadIdx.x, lane = tid & 31, wave = tid >> 5;

    for (int i = tid; i < BATCH * FDIM; i += 256) {       // stage 0: stage x_t -> LDS (bf16)
        int b = i >> 8, f = i & 255;
        xT[f][b] = f2bf(x[((size_t)b * T_STEPS + t) * FDIM + f]);
    }
    __syncthreads();

#pragma unroll                                            // stage 1: Y[m] = Pm @ x_t (WMMA)
    for (int i = 0; i < 8; ++i) {
        int tile = wave * 8 + i;                          // 64 tiles: 2 mats * 2 mtiles * 16 ntiles
        int mat = tile >> 5, r = tile & 31;
        int mtile = r >> 4, nt = r & 15;
        v16bf a = load_A(Pbf, mtile * 16, 64, mat * 32, lane);
        v16bf b = load_B_T(&xT[0][0], nt * 16, BATCH, lane);
        v8f acc = {0.f, 0.f, 0.f, 0.f, 0.f, 0.f, 0.f, 0.f};
        acc = wmma_bf16(a, b, acc);
        int n = nt * 16 + (lane & 15);
        int m0 = mtile * 16 + (lane >> 4) * 8;
#pragma unroll
        for (int rr = 0; rr < 8; ++rr) Y[mat][m0 + rr][n] = f2bf(acc[rr]);
    }
    __syncthreads();

    for (int i = 0; i < 24; ++i) {                        // stage 2: PAX/PBX = Y @ W (WMMA)
        int tile = wave * 24 + i;                         // 192 tiles: 2 mats * 96 gate tiles
        int mat = tile / NTILES, tloc = tile % NTILES;
        int nt = tloc >> 1, mtile = tloc & 1;
        v8f acc = {0.f, 0.f, 0.f, 0.f, 0.f, 0.f, 0.f, 0.f};
#pragma unroll
        for (int ks = 0; ks < KSTEPS; ++ks) {
            v16bf a = load_A(&Y[mat][0][0], mtile * 16, FDIM, ks * 32, lane);
            v16bf b = load_B_pack(Wpack, ks * NT_G + nt, lane);
            acc = wmma_bf16(a, b, acc);
        }
        // tile-packed fragment store: one contiguous 16B vector per lane
        __bf16* out = (mat ? PBX : PAX) + (size_t)t * PAX_ELTS_PER_T
                      + ((size_t)tloc * 32 + lane) * 8;
        v8bf o;
#pragma unroll
        for (int rr = 0; rr < 8; ++rr) o[rr] = f2bf(acc[rr]);
        *(v8bf*)out = o;
    }
}

// ---------- serial recurrence: one persistent workgroup (32 wave32s) ----------
__global__ __launch_bounds__(1024, 1) void recurrent_kernel(
    const __bf16* __restrict__ Pbf, const __bf16* __restrict__ Upack,
    const __bf16* __restrict__ PAX, const __bf16* __restrict__ PBX,
    const float* __restrict__ rx, __bf16* __restrict__ HmT,
    const float* __restrict__ bias, const float* __restrict__ Pr,
    const float* __restrict__ Ur, const float* __restrict__ Wout,
    const float* __restrict__ bout, float* __restrict__ out) {
    __shared__ __align__(64) __bf16 hT[HDIM][BATCH];          // h transposed
    __shared__ __align__(64) __bf16 hrecS[HDIM][BATCH];       // staged Hm[idx] (transposed)
    __shared__ __align__(64) __bf16 heff[BATCH][HDIM];        // P1@h + P2@Hm[idx], row-major
    __shared__ __align__(64) __bf16 gatesF[NTILES][32][8];    // activated i|f|g, fragment order
    __shared__ __align__(64) __bf16 PbfS[BATCH][64];
    __shared__ float red[1024];
    __shared__ float red2[BATCH];
    __shared__ float UrS[HDIM];
    __shared__ float PrS[BATCH];
    __shared__ float biasS[GCOLS];
    __shared__ int   idxS;

    int tid = threadIdx.x, lane = tid & 31, wave = tid >> 5;
    // fragment ownership for the cell update: thread = (i-tile = wave, lane)
    int nt_own = wave >> 1, mt_own = wave & 1;
    int n_own  = nt_own * 16 + (lane & 15);                   // hidden column (0..255)
    int m0_own = mt_own * 16 + (lane >> 4) * 8;               // first of 8 batch rows
    float c_reg[8];
#pragma unroll
    for (int j = 0; j < 8; ++j) c_reg[j] = 0.f;

    // init: h=0 in LDS, HmT slot 0 = 0, stage constants
    for (int i = tid; i < HDIM * BATCH; i += 1024) {
        ((__bf16*)hT)[i] = f2bf(0.f);
        HmT[i]           = f2bf(0.f);
    }
    for (int i = tid; i < BATCH * 64; i += 1024) ((__bf16*)PbfS)[i] = Pbf[i];
    for (int i = tid; i < HDIM; i += 1024) UrS[i] = Ur[i];
    if (tid < BATCH) PrS[tid] = Pr[tid];
    for (int i = tid; i < GCOLS; i += 1024) biasS[i] = bias[i];
    __syncthreads();

    for (int t = 0; t < T_STEPS; ++t) {
        // (a) recall gate = sigmoid(sum_b Pr[b]*(r_x[t,b] + h[b,:].Ur)); idx = recall slot
        // column-parallel: thread = (col = tid>>2, batch-quarter q = tid&3)
        {
            int col = tid >> 2, q = tid & 3;
            v8bf hv = *(const v8bf*)&hT[col][q * 8];
            float s = 0.f;
#pragma unroll
            for (int k = 0; k < 8; ++k) s += bf2f(hv[k]) * PrS[q * 8 + k];
            red[tid] = UrS[col] * s;
        }
        __syncthreads();
        if (tid < 32) {
            float s = 0.f;
#pragma unroll
            for (int k = 0; k < 32; ++k) s += red[tid * 32 + k];
            red2[tid] = s;
        }
        __syncthreads();
        if (tid == 0) {
            float tot = 0.f;
#pragma unroll
            for (int k = 0; k < BATCH; ++k) tot += red2[k] + PrS[k] * rx[t * BATCH + k];
            float gate = sigmoidf_(tot);
            int idx = (int)roundf((float)t * gate);
            int hi = (t - 1 > 0) ? (t - 1) : 0;
            if (idx < 0) idx = 0;
            if (idx > hi) idx = hi;
            if (t < 32) idx = t;   // below recall threshold: self (HmT[t] holds carry h)
            idxS = idx;
        }
        __syncthreads();
        int idx = idxS;

        // stage Hm[idx] (16KB) into LDS: Tensor Data Mover if available, else async copies
#ifdef HAVE_TDM
        if (wave == 0) {
            tdm_load_1d(HmT + (size_t)idx * HDIM * BATCH, &hrecS[0][0],
                        (uint32_t)(HDIM * BATCH * 2));
            wait_tensor0();
        }
#else
        {
            const __bf16* gsrc = HmT + (size_t)idx * HDIM * BATCH + (size_t)tid * 8;
            async_load_b128_to_lds(gsrc, &((__bf16*)hrecS)[(size_t)tid * 8]);
            wait_async0();
        }
#endif
        __syncthreads();

        // (b) heff = P1@h + P2@Hm[idx] : one 16x16 tile per wave, 2 WMMAs each
        {
            int mtile = wave & 1, nt = wave >> 1;          // nt in 0..15
            v8f acc = {0.f, 0.f, 0.f, 0.f, 0.f, 0.f, 0.f, 0.f};
            v16bf a1 = load_A(&PbfS[0][0], mtile * 16, 64, 0, lane);
            v16bf b1 = load_B_T(&hT[0][0], nt * 16, BATCH, lane);
            acc = wmma_bf16(a1, b1, acc);
            v16bf a2 = load_A(&PbfS[0][0], mtile * 16, 64, 32, lane);
            v16bf b2 = load_B_T(&hrecS[0][0], nt * 16, BATCH, lane);
            acc = wmma_bf16(a2, b2, acc);
            int n = nt * 16 + (lane & 15);
            int m0 = mtile * 16 + (lane >> 4) * 8;
#pragma unroll
            for (int rr = 0; rr < 8; ++rr) heff[m0 + rr][n] = f2bf(acc[rr]);
        }
        __syncthreads();

        // (c) gates = heff@U + PAX[t] + PBX[idx] + bias; activate; 3 tiles/wave
        if (t + 1 < T_STEPS) {   // warm L2 for next step's precomputed activations
            __builtin_prefetch(PAX + (size_t)(t + 1) * PAX_ELTS_PER_T + (size_t)wave * 768, 0, 1);
            __builtin_prefetch(PBX + (size_t)(t + 1) * PAX_ELTS_PER_T + (size_t)wave * 768, 0, 1);
        }
        const __bf16* paxt = PAX + (size_t)t   * PAX_ELTS_PER_T;
        const __bf16* pbxi = PBX + (size_t)idx * PAX_ELTS_PER_T;
#pragma unroll
        for (int i = 0; i < 3; ++i) {
            int tile = wave * 3 + i;                        // 96 tiles = 48 ntiles * 2 mtiles
            int nt = tile >> 1, mtile = tile & 1;
            v8f acc = {0.f, 0.f, 0.f, 0.f, 0.f, 0.f, 0.f, 0.f};
#pragma unroll
            for (int ks = 0; ks < KSTEPS; ++ks) {
                v16bf a = load_A(&heff[0][0], mtile * 16, HDIM, ks * 32, lane);
                v16bf b = load_B_pack(Upack, ks * NT_G + nt, lane);
                acc = wmma_bf16(a, b, acc);
            }
            // tile-packed fragment loads: one contiguous 16B vector per lane each
            v8bf pa = *(const v8bf*)(paxt + ((size_t)tile * 32 + lane) * 8);
            v8bf pb = *(const v8bf*)(pbxi + ((size_t)tile * 32 + lane) * 8);
            int n   = nt * 16 + (lane & 15);
            int grp = n >> 8;                               // 0=i, 1=f, 2=g (uniform per tile)
            v8bf o;
#pragma unroll
            for (int rr = 0; rr < 8; ++rr) {
                float v = acc[rr] + bf2f(pa[rr]) + bf2f(pb[rr]) + biasS[n];
                v = (grp == 2) ? tanhf_(v) : sigmoidf_(v);
                o[rr] = f2bf(v);
            }
            *(v8bf*)&gatesF[tile][lane][0] = o;             // one ds_store_b128
        }
        __syncthreads();

        // (d) cell update in fragment space: thread owns i-tile `wave`, f at +32, g at +64
        {
            v8bf iv8 = *(const v8bf*)&gatesF[wave][lane][0];
            v8bf fv8 = *(const v8bf*)&gatesF[wave + 32][lane][0];
            v8bf gv8 = *(const v8bf*)&gatesF[wave + 64][lane][0];
            v8bf hv8;
#pragma unroll
            for (int rr = 0; rr < 8; ++rr) {
                float iv = bf2f(iv8[rr]), fv = bf2f(fv8[rr]), gv = bf2f(gv8[rr]);
                c_reg[rr] = fv * c_reg[rr] + iv * gv;
                hv8[rr] = f2bf(iv * tanhf_(c_reg[rr]));
            }
            *(v8bf*)&hT[n_own][m0_own] = hv8;               // one ds_store_b128 (transposed h)
        }
        __syncthreads();

        // publish h for step t+1: async bulk copy hT (LDS) -> HmT[t+1] (global)
        if (t + 1 < T_STEPS) {
            __bf16* gdst = HmT + (size_t)(t + 1) * HDIM * BATCH + (size_t)tid * 8;
            async_store_b128_from_lds(gdst, &((const __bf16*)hT)[(size_t)tid * 8]);
            wait_async0();
        }
        __syncthreads();
    }

    // out[b] = h[b,:] @ W_out + b_out   (read h from transposed slab; one-shot)
    {
        int b = wave, c0 = lane * 8;
        float part = 0.f;
#pragma unroll
        for (int j = 0; j < 8; ++j) part += bf2f(hT[c0 + j][b]) * Wout[c0 + j];
        red[tid] = part;
    }
    __syncthreads();
    if (tid < BATCH) {
        float s = 0.f;
#pragma unroll
        for (int k = 0; k < 32; ++k) s += red[tid * 32 + k];
        out[tid] = s + bout[0];
    }
}

// ---------- host ----------
extern "C" void kernel_launch(void* const* d_in, const int* in_sizes, int n_in,
                              void* d_out, int out_size, void* d_ws, size_t ws_size,
                              hipStream_t stream) {
    (void)in_sizes; (void)n_in; (void)out_size; (void)ws_size;
    const float* x    = (const float*)d_in[0];
    const float* W    = (const float*)d_in[1];
    const float* U    = (const float*)d_in[2];
    const float* P    = (const float*)d_in[3];
    const float* Bb   = (const float*)d_in[4];
    const float* Wr   = (const float*)d_in[5];
    const float* Pr   = (const float*)d_in[6];
    const float* Ur   = (const float*)d_in[7];
    const float* Wout = (const float*)d_in[8];
    const float* bout = (const float*)d_in[9];
    float* out = (float*)d_out;

    char* p = (char*)d_ws;
    auto take = [&](size_t bytes) { char* r = p; p += (bytes + 255) & ~(size_t)255; return r; };
    __bf16* Pbf   = (__bf16*)take((size_t)BATCH * 64 * 2);
    __bf16* Wpack = (__bf16*)take((size_t)KSTEPS * NT_G * 32 * 16 * 2);
    __bf16* Upack = (__bf16*)take((size_t)KSTEPS * NT_G * 32 * 16 * 2);
    float*  rxb   = (float*) take((size_t)T_STEPS * BATCH * 4);
    __bf16* PAX   = (__bf16*)take((size_t)T_STEPS * PAX_ELTS_PER_T * 2);
    __bf16* PBX   = (__bf16*)take((size_t)T_STEPS * PAX_ELTS_PER_T * 2);
    __bf16* HmT   = (__bf16*)take((size_t)T_STEPS * HDIM * BATCH * 2);

    convert_p_kernel<<<8, 256, 0, stream>>>(P, Pbf);
    pack_weight_kernel<<<48, 256, 0, stream>>>(W, Wpack);
    pack_weight_kernel<<<48, 256, 0, stream>>>(U, Upack);
    recall_x_kernel<<<128, 256, 0, stream>>>(x, Wr, rxb);
    precompute_kernel<<<T_STEPS, 256, 0, stream>>>(x, Pbf, Wpack, PAX, PBX);
    recurrent_kernel<<<1, 1024, 0, stream>>>(Pbf, Upack, PAX, PBX, rxb, HmT,
                                             Bb, Pr, Ur, Wout, bout, out);
}
